// GAT_COO_22127671509516
// MI455X (gfx1250) — compile-verified, matrix operand order checked by reading, not run on hardware
//
#include <hip/hip_runtime.h>
#include <hip/hip_bf16.h>
#include <cstdint>

// ---------------- WMMA types (CDNA5 gfx1250, wave32) ----------------
typedef __attribute__((ext_vector_type(16))) __bf16 v16bf;
typedef __attribute__((ext_vector_type(8)))  float  v8f;

union ABf16 {
    v16bf v;
    unsigned short u[16];
    uint4 q[2];
};

// Constants from the reference
#define F_IN    128
#define HC      256      // H*C
#define HID     128
#define NCLASS  10
#define NCPAD   16
#define NEGS    0.2f

__device__ __forceinline__ unsigned short f2bf(float f) {
    unsigned u = __float_as_uint(f);
    unsigned r = (u + 0x7FFFu + ((u >> 16) & 1u)) >> 16;   // RNE
    return (unsigned short)r;
}

__device__ __forceinline__ void atomicMaxFloat(float* addr, float val) {
    if (val >= 0.0f) atomicMax((int*)addr, __float_as_int(val));
    else             atomicMin((unsigned int*)addr, __float_as_uint(val));
}

__device__ __forceinline__ void edge_sd(const long long* __restrict__ ei, int E, int e,
                                        int& s, int& d) {
    if (e < E) { s = (int)ei[e]; d = (int)ei[(size_t)E + e]; }
    else       { s = e - E; d = e - E; }      // implicit self-loops
}

// ---------------- generic fill / bias-init ----------------
__global__ void fill_f32(float* __restrict__ p, float val, size_t n) {
    size_t i = (size_t)blockIdx.x * blockDim.x + threadIdx.x;
    if (i < n) p[i] = val;
}

__global__ void init_bias(float* __restrict__ p, const float* __restrict__ bias,
                          int ncols, size_t n) {
    size_t i = (size_t)blockIdx.x * blockDim.x + threadIdx.x;
    if (i < n) p[i] = bias[i % ncols];
}

// ---------------- f32 -> bf16 convert (n multiple of 4) ----------------
__global__ void conv_bf16(const float* __restrict__ src, unsigned short* __restrict__ dst,
                          size_t n) {
    size_t i = ((size_t)blockIdx.x * blockDim.x + threadIdx.x) * 4;
    if (i >= n) return;
    float4 f = *(const float4*)(src + i);
    dst[i + 0] = f2bf(f.x);
    dst[i + 1] = f2bf(f.y);
    dst[i + 2] = f2bf(f.z);
    dst[i + 3] = f2bf(f.w);
}

// ---------------- weight transpose + bf16 convert (zero-padded) ----------------
// W: [K, Nc] f32 row-major  ->  Wt: [NcPad, K] bf16 row-major
__global__ void wtrans_bf16(const float* __restrict__ W, unsigned short* __restrict__ Wt,
                            int K, int Nc, int NcPad) {
    int i = blockIdx.x * blockDim.x + threadIdx.x;
    int total = NcPad * K;
    if (i >= total) return;
    int n = i / K, k = i - n * K;
    Wt[(size_t)n * K + k] = (n < Nc) ? f2bf(W[(size_t)k * Nc + n]) : (unsigned short)0;
}

// ---------------- WMMA GEMM: C[M,Ncols] = A[M,K](bf16) x Bt[Ncols,K](bf16)^T --------
// One 16x16 M-tile x NT N-tiles per wave; 8 waves (256 threads) per block.
// A fragment loaded once per kk and reused across NT WMMAs.
// Per-lane fragment halves are 8 consecutive K bf16 -> single b128 loads.
template<int K, int NT>
__global__ void gemm_bf16_wmma(const unsigned short* __restrict__ A,
                               const unsigned short* __restrict__ Bt,
                               float* __restrict__ C,
                               int Ncols, int grpN, int totalGrp) {
    int wave = threadIdx.x >> 5;
    int grp  = blockIdx.x * (blockDim.x >> 5) + wave;
    if (grp >= totalGrp) return;
    int tm = grp / grpN;
    int tg = grp - tm * grpN;
    int lane  = threadIdx.x & 31;
    int lrow  = lane & 15;
    int klane = (lane >> 4) << 3;              // lanes 16-31 hold K+8 / K+24 halves

    const unsigned short* Ap  = A  + (size_t)(tm * 16 + lrow) * K + klane;
    const unsigned short* Bp0 = Bt + (size_t)(tg * NT * 16 + lrow) * K + klane;

    v8f acc[NT];
#pragma unroll
    for (int t = 0; t < NT; ++t) acc[t] = {};

#pragma unroll
    for (int kk = 0; kk < K; kk += 32) {
        ABf16 a;
        a.q[0] = *(const uint4*)(Ap + kk);          // K = kk+klane+0..7
        a.q[1] = *(const uint4*)(Ap + kk + 16);     // K = kk+klane+16..23
#pragma unroll
        for (int t = 0; t < NT; ++t) {
            const unsigned short* Bp = Bp0 + (size_t)t * 16 * K;
            ABf16 b;
            b.q[0] = *(const uint4*)(Bp + kk);
            b.q[1] = *(const uint4*)(Bp + kk + 16);
            acc[t] = __builtin_amdgcn_wmma_f32_16x16x32_bf16(
                false, a.v, false, b.v, (short)0, acc[t], false, false);
        }
    }

    int rowC0 = tm * 16 + ((lane >> 4) << 3);
#pragma unroll
    for (int t = 0; t < NT; ++t) {
        int colC = (tg * NT + t) * 16 + lrow;
#pragma unroll
        for (int v = 0; v < 8; ++v)
            C[(size_t)(rowC0 + v) * Ncols + colC] = acc[t][v];
    }
}

// ---------------- attention coefficients: a_s/a_d [N,4] ----------------
__global__ void att_kernel(const float* __restrict__ h,
                           const float* __restrict__ att_src,
                           const float* __restrict__ att_dst,
                           float* __restrict__ a_s, float* __restrict__ a_d, int Nn) {
    int t = blockIdx.x * blockDim.x + threadIdx.x;
    if (t >= Nn * 4) return;
    int n = t >> 2, hd = t & 3;
    const float* hp = h + (size_t)n * HC + hd * 64;
    const float* as = att_src + hd * 64;
    const float* ad = att_dst + hd * 64;
    float ss = 0.f, sd = 0.f;
#pragma unroll
    for (int c = 0; c < 64; c += 4) {
        float4 hv = *(const float4*)(hp + c);
        float4 sv = *(const float4*)(as + c);
        float4 dv = *(const float4*)(ad + c);
        ss += hv.x * sv.x + hv.y * sv.y + hv.z * sv.z + hv.w * sv.w;
        sd += hv.x * dv.x + hv.y * dv.y + hv.z * dv.z + hv.w * dv.w;
    }
    a_s[t] = ss;
    a_d[t] = sd;
}

// ---------------- segment max over incoming edges ----------------
__global__ void edge_max_kernel(const long long* __restrict__ ei, int E, int E2,
                                const float* __restrict__ a_s, const float* __restrict__ a_d,
                                float* __restrict__ m) {
    int e = blockIdx.x * blockDim.x + threadIdx.x;
    if (e >= E2) return;
    int s, d; edge_sd(ei, E, e, s, d);
    float4 as = *(const float4*)(a_s + (size_t)s * 4);
    float4 ad = *(const float4*)(a_d + (size_t)d * 4);
    float v0 = as.x + ad.x, v1 = as.y + ad.y, v2 = as.z + ad.z, v3 = as.w + ad.w;
    v0 = (v0 > 0.f) ? v0 : NEGS * v0;
    v1 = (v1 > 0.f) ? v1 : NEGS * v1;
    v2 = (v2 > 0.f) ? v2 : NEGS * v2;
    v3 = (v3 > 0.f) ? v3 : NEGS * v3;
    atomicMaxFloat(&m[d * 4 + 0], v0);
    atomicMaxFloat(&m[d * 4 + 1], v1);
    atomicMaxFloat(&m[d * 4 + 2], v2);
    atomicMaxFloat(&m[d * 4 + 3], v3);
}

// ---------------- exp-sum denominator + degree ----------------
__global__ void edge_sum_kernel(const long long* __restrict__ ei, int E, int E2,
                                const float* __restrict__ a_s, const float* __restrict__ a_d,
                                const float* __restrict__ m,
                                float* __restrict__ denom, float* __restrict__ deg) {
    int e = blockIdx.x * blockDim.x + threadIdx.x;
    if (e >= E2) return;
    int s, d; edge_sd(ei, E, e, s, d);
    float4 as = *(const float4*)(a_s + (size_t)s * 4);
    float4 ad = *(const float4*)(a_d + (size_t)d * 4);
    float4 mm = *(const float4*)(m + (size_t)d * 4);
    float v0 = as.x + ad.x, v1 = as.y + ad.y, v2 = as.z + ad.z, v3 = as.w + ad.w;
    v0 = (v0 > 0.f) ? v0 : NEGS * v0;
    v1 = (v1 > 0.f) ? v1 : NEGS * v1;
    v2 = (v2 > 0.f) ? v2 : NEGS * v2;
    v3 = (v3 > 0.f) ? v3 : NEGS * v3;
    atomicAdd(&denom[d * 4 + 0], __expf(v0 - mm.x));
    atomicAdd(&denom[d * 4 + 1], __expf(v1 - mm.y));
    atomicAdd(&denom[d * 4 + 2], __expf(v2 - mm.z));
    atomicAdd(&denom[d * 4 + 3], __expf(v3 - mm.w));
    atomicAdd(&deg[d], 1.0f);
}

// ---------------- weighted aggregation: thread = (edge, 4 channels) ----------------
__global__ void edge_agg_kernel(const long long* __restrict__ ei, int E, int E2,
                                const float* __restrict__ a_s, const float* __restrict__ a_d,
                                const float* __restrict__ m, const float* __restrict__ denom,
                                const float* __restrict__ h, float* __restrict__ agg) {
    long long idx = (long long)blockIdx.x * blockDim.x + threadIdx.x;
    int e = (int)(idx >> 6);                   // 64 threads per edge
    if (e >= E2) return;
    int c4 = ((int)idx & 63) * 4;              // channel base 0..252
    int hd = c4 >> 6;
    int s, d; edge_sd(ei, E, e, s, d);
    float v = a_s[s * 4 + hd] + a_d[d * 4 + hd];
    v = (v > 0.f) ? v : NEGS * v;
    float alpha = __expf(v - m[d * 4 + hd]) / denom[d * 4 + hd];
    float4 hv = *(const float4*)(h + (size_t)s * HC + c4);
    float* ap = agg + (size_t)d * HC + c4;
    atomicAdd(ap + 0, alpha * hv.x);
    atomicAdd(ap + 1, alpha * hv.y);
    atomicAdd(ap + 2, alpha * hv.z);
    atomicAdd(ap + 3, alpha * hv.w);
}

// ---------------- xg = elu(agg + bg) -> bf16 directly ----------------
__global__ void elu_bias_bf16_kernel(const float* __restrict__ agg, const float* __restrict__ bg,
                                     unsigned short* __restrict__ xgb, size_t n) {
    size_t i = (size_t)blockIdx.x * blockDim.x + threadIdx.x;
    if (i >= n) return;
    float v = agg[i] + bg[i % HC];
    v = (v > 0.f) ? v : (__expf(v) - 1.0f);
    xgb[i] = f2bf(v);
}

// ---------------- dinv = deg > 0 ? rsqrt(deg) : 0 ----------------
__global__ void dinv_kernel(const float* __restrict__ deg, float* __restrict__ dinv, int Nn) {
    int i = blockIdx.x * blockDim.x + threadIdx.x;
    if (i >= Nn) return;
    float dg = deg[i];
    dinv[i] = (dg > 0.f) ? rsqrtf(dg) : 0.f;
}

// ---------------- GCN sym-norm SpMM scatter: thread = (edge, 4 channels) ------------
__global__ void gcn_agg_kernel(const long long* __restrict__ ei, int E, int E2,
                               const float* __restrict__ dinv, const float* __restrict__ xw,
                               float* __restrict__ xc) {
    long long idx = (long long)blockIdx.x * blockDim.x + threadIdx.x;
    int e = (int)(idx >> 5);                   // 32 threads per edge
    if (e >= E2) return;
    int c4 = ((int)idx & 31) * 4;              // channel base 0..124
    int s, d; edge_sd(ei, E, e, s, d);
    float norm = dinv[s] * dinv[d];
    float4 w = *(const float4*)(xw + (size_t)s * HID + c4);
    float* xp = xc + (size_t)d * HID + c4;
    atomicAdd(xp + 0, norm * w.x);
    atomicAdd(xp + 1, norm * w.y);
    atomicAdd(xp + 2, norm * w.z);
    atomicAdd(xp + 3, norm * w.w);
}

// ---------------- relu -> bf16 ----------------
__global__ void relu_bf16_kernel(const float* __restrict__ xc, unsigned short* __restrict__ xcb,
                                 size_t n) {
    size_t i = (size_t)blockIdx.x * blockDim.x + threadIdx.x;
    if (i < n) xcb[i] = f2bf(fmaxf(xc[i], 0.f));
}

// ---------------- bias + log_softmax over 10 classes ----------------
__global__ void logsoftmax_kernel(const float* __restrict__ logits, const float* __restrict__ bl,
                                  float* __restrict__ out, int Nn) {
    int n = blockIdx.x * blockDim.x + threadIdx.x;
    if (n >= Nn) return;
    float v[NCLASS];
    float mx = -3.4e38f;
#pragma unroll
    for (int j = 0; j < NCLASS; ++j) {
        v[j] = logits[(size_t)n * NCPAD + j] + bl[j];
        mx = fmaxf(mx, v[j]);
    }
    float sum = 0.f;
#pragma unroll
    for (int j = 0; j < NCLASS; ++j) sum += __expf(v[j] - mx);
    float ls = mx + __logf(sum);
#pragma unroll
    for (int j = 0; j < NCLASS; ++j) out[(size_t)n * NCLASS + j] = v[j] - ls;
}

// =====================================================================
static inline size_t align256(size_t x) { return (x + 255) & ~(size_t)255; }

extern "C" void kernel_launch(void* const* d_in, const int* in_sizes, int n_in,
                              void* d_out, int out_size, void* d_ws, size_t ws_size,
                              hipStream_t stream) {
    const float* x       = (const float*)d_in[0];
    // d_in[1] = edge_attr: ignored (edge_dim=None)
    const float* Wg      = (const float*)d_in[2];
    const float* att_src = (const float*)d_in[3];
    const float* att_dst = (const float*)d_in[4];
    const float* bg      = (const float*)d_in[5];
    const float* Wc      = (const float*)d_in[6];
    const float* bc      = (const float*)d_in[7];
    const float* Wl      = (const float*)d_in[8];
    const float* bl      = (const float*)d_in[9];
    const long long* ei  = (const long long*)d_in[10];   // jnp.int64

    const int Nn = in_sizes[0] / F_IN;     // 50000 (= 3125*16)
    const int E  = in_sizes[10] / 2;       // 800000
    const int E2 = E + Nn;                 // + self loops

    // ---- workspace bump allocation (256B aligned) ----
    char* base = (char*)d_ws;
    size_t off = 0;
    auto alloc = [&](size_t bytes) { size_t o = off; off = align256(off + bytes); return (void*)(base + o); };

    float* h      = (float*)alloc((size_t)Nn * HC  * 4);
    float* agg    = (float*)alloc((size_t)Nn * HC  * 4);
    float* xw     = (float*)alloc((size_t)Nn * HID * 4);
    float* xc     = (float*)alloc((size_t)Nn * HID * 4);
    float* logits = (float*)alloc((size_t)Nn * NCPAD * 4);
    float* a_s    = (float*)alloc((size_t)Nn * 4 * 4);
    float* a_d    = (float*)alloc((size_t)Nn * 4 * 4);
    float* m      = (float*)alloc((size_t)Nn * 4 * 4);
    float* denom  = (float*)alloc((size_t)Nn * 4 * 4);
    float* deg    = (float*)alloc((size_t)Nn * 4);
    float* dinv   = (float*)alloc((size_t)Nn * 4);
    unsigned short* xB   = (unsigned short*)alloc((size_t)Nn * F_IN * 2);
    unsigned short* xgB  = (unsigned short*)alloc((size_t)Nn * HC   * 2);
    unsigned short* xcB  = (unsigned short*)alloc((size_t)Nn * HID  * 2);
    unsigned short* WgT  = (unsigned short*)alloc((size_t)HC    * F_IN * 2);
    unsigned short* WcT  = (unsigned short*)alloc((size_t)HID   * HC   * 2);
    unsigned short* WlT  = (unsigned short*)alloc((size_t)NCPAD * HID  * 2);
    (void)ws_size;

    const int T = 256;
    auto cdiv = [](size_t a, size_t b) { return (int)((a + b - 1) / b); };

    // ---- weight transpose/convert + activation convert ----
    wtrans_bf16<<<cdiv((size_t)HC * F_IN,  T), T, 0, stream>>>(Wg, WgT, F_IN, HC,     HC);
    wtrans_bf16<<<cdiv((size_t)HID * HC,   T), T, 0, stream>>>(Wc, WcT, HC,   HID,    HID);
    wtrans_bf16<<<cdiv((size_t)NCPAD*HID,  T), T, 0, stream>>>(Wl, WlT, HID,  NCLASS, NCPAD);
    conv_bf16<<<cdiv((size_t)Nn * F_IN / 4, T), T, 0, stream>>>(x, xB, (size_t)Nn * F_IN);

    // ---- GEMM1: h = x @ Wg   [N,128]x[128,256]  (NT=4 -> 4 n-groups) ----
    {
        int grpN = (HC / 16) / 4, total = (Nn / 16) * grpN;
        gemm_bf16_wmma<F_IN, 4><<<cdiv(total, 8), T, 0, stream>>>(xB, WgT, h, HC, grpN, total);
    }

    // ---- attention coefficients ----
    att_kernel<<<cdiv((size_t)Nn * 4, T), T, 0, stream>>>(h, att_src, att_dst, a_s, a_d, Nn);

    // ---- init accumulators ----
    fill_f32<<<cdiv((size_t)Nn * 4,  T), T, 0, stream>>>(m,     -3.4e38f, (size_t)Nn * 4);
    fill_f32<<<cdiv((size_t)Nn * 4,  T), T, 0, stream>>>(denom, 0.f,      (size_t)Nn * 4);
    fill_f32<<<cdiv((size_t)Nn,      T), T, 0, stream>>>(deg,   0.f,      (size_t)Nn);
    fill_f32<<<cdiv((size_t)Nn * HC, T), T, 0, stream>>>(agg,   0.f,      (size_t)Nn * HC);

    // ---- segment softmax (3 streaming edge passes) ----
    edge_max_kernel<<<cdiv(E2, T), T, 0, stream>>>(ei, E, E2, a_s, a_d, m);
    edge_sum_kernel<<<cdiv(E2, T), T, 0, stream>>>(ei, E, E2, a_s, a_d, m, denom, deg);
    edge_agg_kernel<<<cdiv((size_t)E2 * 64, T), T, 0, stream>>>(ei, E, E2, a_s, a_d, m, denom, h, agg);

    // ---- xg = elu(agg + bg) -> bf16 ----
    elu_bias_bf16_kernel<<<cdiv((size_t)Nn * HC, T), T, 0, stream>>>(agg, bg, xgB, (size_t)Nn * HC);

    // ---- GEMM2: xw = xg @ Wc   [N,256]x[256,128] ----
    {
        int grpN = (HID / 16) / 4, total = (Nn / 16) * grpN;
        gemm_bf16_wmma<HC, 4><<<cdiv(total, 8), T, 0, stream>>>(xgB, WcT, xw, HID, grpN, total);
    }

    // ---- GCN: xc = sum(norm * xw[src]) + bc, relu -> bf16 ----
    dinv_kernel<<<cdiv((size_t)Nn, T), T, 0, stream>>>(deg, dinv, Nn);
    init_bias<<<cdiv((size_t)Nn * HID, T), T, 0, stream>>>(xc, bc, HID, (size_t)Nn * HID);
    gcn_agg_kernel<<<cdiv((size_t)E2 * 32, T), T, 0, stream>>>(ei, E, E2, dinv, xw, xc);
    relu_bf16_kernel<<<cdiv((size_t)Nn * HID, T), T, 0, stream>>>(xc, xcB, (size_t)Nn * HID);

    // ---- GEMM3: logits = xc @ Wl (padded to 16 cols, NT=1) ----
    {
        int grpN = 1, total = (Nn / 16);
        gemm_bf16_wmma<HID, 1><<<cdiv(total, 8), T, 0, stream>>>(xcB, WlT, logits, NCPAD, grpN, total);
    }

    // ---- bias + log_softmax -> d_out [N,10] ----
    logsoftmax_kernel<<<cdiv((size_t)Nn, T), T, 0, stream>>>(logits, bl, (float*)d_out, Nn);
}
